// FalconMamba2SSMDecoderLayer_44418551776018
// MI455X (gfx1250) — matrix operand order repaired and checked by reading, not compiled, not run on hardware
//
#include <hip/hip_runtime.h>
#include <hip/hip_bf16.h>

typedef __bf16 bf16_t;
typedef __attribute__((ext_vector_type(16))) __bf16 v16bf;
typedef __attribute__((ext_vector_type(8)))  __bf16 v8bf;
typedef __attribute__((ext_vector_type(8)))  float  v8f;

#define L_SEQ   2048
#define H_DIM   2048
#define D_SSM   4096
#define NH      64
#define PD      64        // head_dim
#define NS      128       // ssm state size
#define KCONV   4
#define CS      256       // chunk size
#define NC      8         // L / CS
#define PROJ_DIM 8512     // 2*D_SSM + 2*N + NH
#define CONV_DIM 4352     // D_SSM + 2*N
#define EPSF    1e-5f

__device__ __forceinline__ bf16_t f2bf(float f) {
  unsigned u = __float_as_uint(f);
  u += 0x7FFFu + ((u >> 16) & 1u);          // round-to-nearest-even
  unsigned short s = (unsigned short)(u >> 16);
  bf16_t b; __builtin_memcpy(&b, &s, 2); return b;
}

__device__ __forceinline__ v16bf mk16(v8bf lo, v8bf hi) {
  v16bf r;
#pragma unroll
  for (int i = 0; i < 8; ++i) { r[i] = lo[i]; r[8 + i] = hi[i]; }
  return r;
}
// A frag: lane holds K [kb..kb+8) and [kb+16..kb+24), kb = k0 + 8*(lane>>4)
__device__ __forceinline__ v16bf load_afrag(const bf16_t* arow, int k0, int lh) {
  const bf16_t* p = arow + k0 + lh * 8;
  return mk16(*(const v8bf*)p, *(const v8bf*)(p + 16));
}
// B frag: lane = output column (row of row-major B), K [kb..kb+8) and [kb+16..kb+24), kb = k0 + 8*(lane>>4)
__device__ __forceinline__ v16bf load_bfrag(const bf16_t* brow, int k0, int lh) {
  const bf16_t* p = brow + k0 + lh * 8;
  return mk16(*(const v8bf*)p, *(const v8bf*)(p + 16));
}
#define WMMA_BF16(a, b, c) \
  __builtin_amdgcn_wmma_f32_16x16x32_bf16(false, (a), false, (b), (short)0, (c), false, false)

// ---------------------------------------------------------------- convert
__global__ void cvt_f32_bf16(const float* __restrict__ in, bf16_t* __restrict__ out, int n) {
  int i = blockIdx.x * 256 + threadIdx.x;
  if (i < n) out[i] = f2bf(in[i]);
}

// ---------------------------------------------------------------- input RMSNorm -> bf16
__global__ void rmsnorm_bf16(const float* __restrict__ x, const float* __restrict__ w,
                             bf16_t* __restrict__ out) {
  int row = blockIdx.x, t = threadIdx.x;
  const float* xr = x + (size_t)row * H_DIM;
  float ss = 0.f;
  for (int d = t; d < H_DIM; d += 256) { float v = xr[d]; ss += v * v; }
  __shared__ float red[256];
  red[t] = ss; __syncthreads();
  for (int o = 128; o > 0; o >>= 1) { if (t < o) red[t] += red[t + o]; __syncthreads(); }
  float rms = rsqrtf(red[0] / (float)H_DIM + EPSF);
  bf16_t* orow = out + (size_t)row * H_DIM;
  for (int d = t; d < H_DIM; d += 256) orow[d] = f2bf(xr[d] * rms * w[d]);
}

// ---------------------------------------------------------------- generic C = A * B^T (bf16 in, f32 acc)
// A: M x K row-major (lda), B: N x K row-major (ldb), C: M x N (ldc). Batched via blockIdx.z.
// Wave computes 32x64 (2 A-frags x 4 B-frags = 8 WMMA / K-step). Block of 256 threads = 8 waves
// stacked in M (256 rows per block). launch_bounds(256,1): budget registers for 1 block/EU so
// accumulators and fragments stay in VGPRs (no scratch spills); latency hidden by load clauses.
__global__ void __launch_bounds__(256, 1)
gemm_bf16_abt(const bf16_t* __restrict__ A, const bf16_t* __restrict__ B,
              float* __restrict__ C,
              int M, int N, int K, int lda, int ldb, int ldc,
              long sA, long sB, long sC,
              const float* __restrict__ resid, int ldr) {
  int wv = threadIdx.x >> 5;
  int lane = threadIdx.x & 31;
  int lh = lane >> 4, ll = lane & 15;
  int mb = (blockIdx.y * 8 + wv) * 32;
  int nb = blockIdx.x * 64;
  if (mb >= M || nb >= N) return;
  int bz = blockIdx.z;
  const bf16_t* Ab = A + (size_t)bz * sA;
  const bf16_t* Bb = B + (size_t)bz * sB;
  float* Cb = C + (size_t)bz * sC;

  const bf16_t* ar0 = Ab + (size_t)(mb + ll) * lda;
  const bf16_t* ar1 = Ab + (size_t)(mb + 16 + ll) * lda;
  const bf16_t* br0 = Bb + (size_t)(nb + ll) * ldb;
  const bf16_t* br1 = br0 + (size_t)16 * ldb;
  const bf16_t* br2 = br0 + (size_t)32 * ldb;
  const bf16_t* br3 = br0 + (size_t)48 * ldb;

  v8f acc[2][4];
#pragma unroll
  for (int mt = 0; mt < 2; ++mt)
#pragma unroll
    for (int f = 0; f < 4; ++f)
#pragma unroll
      for (int r = 0; r < 8; ++r) acc[mt][f][r] = 0.f;

  for (int k0 = 0; k0 < K; k0 += 32) {
    v16bf a0 = load_afrag(ar0, k0, lh);
    v16bf a1 = load_afrag(ar1, k0, lh);
    v16bf b0 = load_bfrag(br0, k0, lh);
    v16bf b1 = load_bfrag(br1, k0, lh);
    v16bf b2 = load_bfrag(br2, k0, lh);
    v16bf b3 = load_bfrag(br3, k0, lh);

    acc[0][0] = WMMA_BF16(a0, b0, acc[0][0]);
    acc[0][1] = WMMA_BF16(a0, b1, acc[0][1]);
    acc[0][2] = WMMA_BF16(a0, b2, acc[0][2]);
    acc[0][3] = WMMA_BF16(a0, b3, acc[0][3]);
    acc[1][0] = WMMA_BF16(a1, b0, acc[1][0]);
    acc[1][1] = WMMA_BF16(a1, b1, acc[1][1]);
    acc[1][2] = WMMA_BF16(a1, b2, acc[1][2]);
    acc[1][3] = WMMA_BF16(a1, b3, acc[1][3]);
  }

#pragma unroll
  for (int mt = 0; mt < 2; ++mt)
#pragma unroll
    for (int f = 0; f < 4; ++f) {
      int col = nb + f * 16 + ll;
#pragma unroll
      for (int r = 0; r < 8; ++r) {
        int m = mb + mt * 16 + r + lh * 8;
        float v = acc[mt][f][r];
        if (resid) v += resid[(size_t)m * ldr + col];
        Cb[(size_t)m * ldc + col] = v;
      }
    }
}

// ---------------------------------------------------------------- causal depthwise conv + SiLU
__global__ void conv_silu(const float* __restrict__ proj, const float* __restrict__ cw,
                          const float* __restrict__ cb, float* __restrict__ xc,
                          bf16_t* __restrict__ xsb, bf16_t* __restrict__ Bmb,
                          bf16_t* __restrict__ Cmb) {
  int idx = blockIdx.x * 256 + threadIdx.x;
  if (idx >= L_SEQ * CONV_DIM) return;
  int c = idx % CONV_DIM;
  int l = idx / CONV_DIM;
  float a = cb[c];
#pragma unroll
  for (int k = 0; k < KCONV; ++k) {
    int ls = l + k - (KCONV - 1);
    if (ls >= 0) a += proj[(size_t)ls * PROJ_DIM + D_SSM + c] * cw[c * KCONV + k];
  }
  float s = a / (1.f + __expf(-a));            // SiLU
  xc[(size_t)l * CONV_DIM + c] = s;
  if (c < D_SSM)             xsb[(size_t)l * D_SSM + c] = f2bf(s);
  else if (c < D_SSM + NS)   Bmb[(size_t)l * NS + (c - D_SSM)] = f2bf(s);
  else                       Cmb[(size_t)l * NS + (c - D_SSM - NS)] = f2bf(s);
}

// ---------------------------------------------------------------- dt = softplus(dt + bias)
__global__ void dt_softplus(const float* __restrict__ proj, const float* __restrict__ dtb,
                            float* __restrict__ dts) {
  int idx = blockIdx.x * 256 + threadIdx.x;
  if (idx >= L_SEQ * NH) return;
  int h = idx % NH, l = idx / NH;
  float v = proj[(size_t)l * PROJ_DIM + 2 * D_SSM + 2 * NS + h] + dtb[h];
  dts[idx] = (v > 20.f) ? v : log1pf(__expf(v));
}

// ---------------------------------------------------------------- per-(chunk,head) cumsum of dA
__global__ void dA_cumsum(const float* __restrict__ dts, const float* __restrict__ A_log,
                          float* __restrict__ dAc) {
  int bid = blockIdx.x;                    // c*NH + h
  int h = bid & (NH - 1);
  int c = bid / NH;
  int t = threadIdx.x;
  float Ah = -__expf(A_log[h]);
  float v = dts[(size_t)(c * CS + t) * NH + h] * Ah;
  __shared__ float sb[CS];
  sb[t] = v; __syncthreads();
  for (int o = 1; o < CS; o <<= 1) {
    float u = (t >= o) ? sb[t - o] : 0.f;
    __syncthreads();
    sb[t] += u;
    __syncthreads();
  }
  dAc[(size_t)bid * CS + t] = sb[t];
}

// ---------------------------------------------------------------- Y_diag = (CB .* L .* dt) @ xs  (+ D*xs)
// xs chunk-head tile (256x64 bf16 = 32KB) staged in LDS, shared by all 8 waves.
__global__ void __launch_bounds__(256, 1)
ydiag_wmma(const float* __restrict__ CBuf, const float* __restrict__ dAc,
           const float* __restrict__ dts, const bf16_t* __restrict__ xsb,
           const float* __restrict__ xc, const float* __restrict__ Dv,
           float* __restrict__ y) {
  int h = blockIdx.x, c = blockIdx.y;
  int t = threadIdx.x;
  __shared__ float sA[CS], sdt[CS];
  __shared__ __align__(16) bf16_t sxs[CS * PD];
  sA[t]  = dAc[(size_t)(c * NH + h) * CS + t];
  sdt[t] = dts[(size_t)(c * CS + t) * NH + h];
  // cooperative coalesced load of xs[c*CS .. +256)[h*PD .. +64) into LDS
  for (int j = t; j < CS * PD / 8; j += 256) {        // 2048 x v8bf
    int row = j >> 3;
    int cc = (j & 7) * 8;
    *(v8bf*)&sxs[row * PD + cc] =
        *(const v8bf*)&xsb[(size_t)(c * CS + row) * D_SSM + h * PD + cc];
  }
  __syncthreads();
  int wv = t >> 5, lane = t & 31, lh = lane >> 4, ll = lane & 15;

  v8f acc[2][4];
#pragma unroll
  for (int a = 0; a < 2; ++a)
#pragma unroll
    for (int f = 0; f < 4; ++f)
#pragma unroll
      for (int r = 0; r < 8; ++r) acc[a][f][r] = 0.f;

  int rowTop = wv * 32 + 32;                 // causality: K < rowTop for this wave
  for (int s0 = 0; s0 < rowTop; s0 += 32) {
    // bf16 xs fragments from LDS (B matrix, K=s over rows of the tile)
    v16bf bv[4];
#pragma unroll
    for (int f = 0; f < 4; ++f) {
      int sr = s0 + lh * 8;
      int pc = f * 16 + ll;
#pragma unroll
      for (int i = 0; i < 16; ++i) bv[f][i] = sxs[(sr + i + (i & 8)) * PD + pc];
    }
#pragma unroll
    for (int mt = 0; mt < 2; ++mt) {
      int r0 = wv * 32 + mt * 16;
      if (s0 >= r0 + 16) continue;           // fully above diagonal for this m-tile
      int row = r0 + ll;
      float dal = sA[row];
      const float* cbrow = CBuf + ((size_t)c * CS + row) * CS;
      int kb = s0 + lh * 8;
      v16bf af;
#pragma unroll
      for (int j = 0; j < 8; ++j) {
        int k1 = kb + j, k2 = kb + 16 + j;
        float m1 = (k1 <= row) ? cbrow[k1] * __expf(dal - sA[k1]) * sdt[k1] : 0.f;
        float m2 = (k2 <= row) ? cbrow[k2] * __expf(dal - sA[k2]) * sdt[k2] : 0.f;
        af[j] = f2bf(m1); af[8 + j] = f2bf(m2);
      }
#pragma unroll
      for (int f = 0; f < 4; ++f)
        acc[mt][f] = WMMA_BF16(af, bv[f], acc[mt][f]);
    }
  }
  float Dh = Dv[h];
#pragma unroll
  for (int mt = 0; mt < 2; ++mt)
#pragma unroll
    for (int f = 0; f < 4; ++f)
#pragma unroll
      for (int r = 0; r < 8; ++r) {
        int m = wv * 32 + mt * 16 + r + lh * 8;
        int lg = c * CS + m;
        int pcol = f * 16 + ll;
        float xsv = xc[(size_t)lg * CONV_DIM + h * PD + pcol];
        y[(size_t)lg * D_SSM + h * PD + pcol] = acc[mt][f][r] + Dh * xsv;
      }
}

// ---------------------------------------------------------------- states[c,h,p,n] = sum_l B[l,n]*w[l]*xs[l,p]
__global__ void states_kernel(const float* __restrict__ xc, const float* __restrict__ dAc,
                              const float* __restrict__ dts, float* __restrict__ st) {
  int h = blockIdx.x, c = blockIdx.y;
  int t = threadIdx.x;
  __shared__ float wl[CS];
  __shared__ float xsT[16][PD];
  __shared__ float bT[16][NS];
  const float* da = dAc + (size_t)(c * NH + h) * CS;
  float da_last = da[CS - 1];
  wl[t] = __expf(da_last - da[t]) * dts[(size_t)(c * CS + t) * NH + h];
  __syncthreads();
  float acc[32];
#pragma unroll
  for (int j = 0; j < 32; ++j) acc[j] = 0.f;
  int p = t >> 2, nb = (t & 3) * 32;
  for (int l0 = 0; l0 < CS; l0 += 16) {
    for (int i = t; i < 16 * PD; i += 256) {
      int li = i >> 6, pi = i & 63;
      xsT[li][pi] = xc[(size_t)(c * CS + l0 + li) * CONV_DIM + h * PD + pi];
    }
    for (int i = t; i < 16 * NS; i += 256) {
      int li = i >> 7, ni = i & 127;
      bT[li][ni] = xc[(size_t)(c * CS + l0 + li) * CONV_DIM + D_SSM + ni];
    }
    __syncthreads();
#pragma unroll 4
    for (int li = 0; li < 16; ++li) {
      float a = wl[l0 + li] * xsT[li][p];
#pragma unroll
      for (int j = 0; j < 32; ++j) acc[j] += a * bT[li][nb + j];
    }
    __syncthreads();
  }
  float* so = st + ((size_t)(c * NH + h) * PD + p) * NS + nb;
#pragma unroll
  for (int j = 0; j < 32; ++j) so[j] = acc[j];
}

// ---------------------------------------------------------------- inter-chunk recurrence (emit state BEFORE chunk)
__global__ void chunk_scan(const float* __restrict__ st, const float* __restrict__ dAc,
                           float* __restrict__ pst) {
  int idx = blockIdx.x * 256 + threadIdx.x;       // NH*PD*NS threads
  int h = idx >> 13;                               // / (PD*NS)
  int rem = idx & 8191;
  float carry = 0.f;
#pragma unroll
  for (int c = 0; c < NC; ++c) {
    size_t o = (size_t)(c * NH + h) * (PD * NS) + rem;
    pst[o] = carry;
    float dec = __expf(dAc[(size_t)(c * NH + h) * CS + (CS - 1)]);
    carry = carry * dec + st[o];
  }
}

// ---------------------------------------------------------------- Y_off += exp(dA)[l] * sum_n C[l,n]*prev[p,n]
__global__ void yoff_kernel(const float* __restrict__ xc, const float* __restrict__ dAc,
                            const float* __restrict__ pst, float* __restrict__ y) {
  int h = blockIdx.x, c = blockIdx.y;
  int t = threadIdx.x;                             // = l within chunk
  __shared__ float sprev[PD * NS];                 // 32 KB
  for (int i = t; i < PD * NS; i += 256)
    sprev[i] = pst[(size_t)(c * NH + h) * (PD * NS) + i];
  __syncthreads();
  int lg = c * CS + t;
  float eA = __expf(dAc[(size_t)(c * NH + h) * CS + t]);
  float acc[PD];
#pragma unroll
  for (int p = 0; p < PD; ++p) acc[p] = 0.f;
  const float* crow = xc + (size_t)lg * CONV_DIM + D_SSM + NS;   // Cm f32
  for (int n = 0; n < NS; ++n) {
    float cv = crow[n];
#pragma unroll
    for (int p = 0; p < PD; ++p) acc[p] += cv * sprev[p * NS + n];
  }
  float* yr = y + (size_t)lg * D_SSM + h * PD;
#pragma unroll
  for (int p = 0; p < PD; ++p) yr[p] += eA * acc[p];
}

// ---------------------------------------------------------------- gated RMSNorm -> bf16
__global__ void gate_norm(const float* __restrict__ y, const float* __restrict__ proj,
                          const float* __restrict__ nw, bf16_t* __restrict__ out) {
  int l = blockIdx.x, t = threadIdx.x;
  const float* yr = y + (size_t)l * D_SSM;
  const float* zr = proj + (size_t)l * PROJ_DIM;   // z = proj[:, 0:D_SSM]
  float ss = 0.f;
  for (int d = t; d < D_SSM; d += 256) {
    float z = zr[d];
    float g = yr[d] * (z / (1.f + __expf(-z)));
    ss += g * g;
  }
  __shared__ float red[256];
  red[t] = ss; __syncthreads();
  for (int o = 128; o > 0; o >>= 1) { if (t < o) red[t] += red[t + o]; __syncthreads(); }
  float rms = rsqrtf(red[0] / (float)D_SSM + EPSF);
  bf16_t* orow = out + (size_t)l * D_SSM;
  for (int d = t; d < D_SSM; d += 256) {
    float z = zr[d];
    float g = yr[d] * (z / (1.f + __expf(-z)));
    orow[d] = f2bf(g * rms * nw[d]);
  }
}

// ================================================================ launch
extern "C" void kernel_launch(void* const* d_in, const int* in_sizes, int n_in,
                              void* d_out, int out_size, void* d_ws, size_t ws_size,
                              hipStream_t stream) {
  (void)in_sizes; (void)n_in; (void)out_size; (void)ws_size;
  const float* x     = (const float*)d_in[0];
  const float* rms_w = (const float*)d_in[1];
  const float* W_in  = (const float*)d_in[2];
  const float* cw    = (const float*)d_in[3];
  const float* cb    = (const float*)d_in[4];
  const float* dtb   = (const float*)d_in[5];
  const float* A_log = (const float*)d_in[6];
  const float* Dv    = (const float*)d_in[7];
  const float* nw    = (const float*)d_in[8];
  const float* W_out = (const float*)d_in[9];
  float* out = (float*)d_out;

  char* p = (char*)d_ws;
  auto alloc = [&](size_t bytes) { char* r = p; p += (bytes + 255) & ~(size_t)255; return (void*)r; };
  bf16_t* Winb  = (bf16_t*)alloc((size_t)PROJ_DIM * H_DIM * 2);
  bf16_t* Woutb = (bf16_t*)alloc((size_t)H_DIM * D_SSM * 2);
  bf16_t* hb    = (bf16_t*)alloc((size_t)L_SEQ * H_DIM * 2);
  float*  proj  = (float*) alloc((size_t)L_SEQ * PROJ_DIM * 4);
  float*  xc    = (float*) alloc((size_t)L_SEQ * CONV_DIM * 4);
  float*  dts   = (float*) alloc((size_t)L_SEQ * NH * 4);
  float*  dAc   = (float*) alloc((size_t)NC * NH * CS * 4);
  bf16_t* xsb   = (bf16_t*)alloc((size_t)L_SEQ * D_SSM * 2);
  bf16_t* Bmb   = (bf16_t*)alloc((size_t)L_SEQ * NS * 2);
  bf16_t* Cmb   = (bf16_t*)alloc((size_t)L_SEQ * NS * 2);
  float*  CBuf  = (float*) alloc((size_t)NC * CS * CS * 4);
  float*  st    = (float*) alloc((size_t)NC * NH * PD * NS * 4);
  float*  pst   = (float*) alloc((size_t)NC * NH * PD * NS * 4);
  float*  ybuf  = (float*) alloc((size_t)L_SEQ * D_SSM * 4);
  bf16_t* ynb   = (bf16_t*)alloc((size_t)L_SEQ * D_SSM * 2);

  cvt_f32_bf16<<<(PROJ_DIM * H_DIM + 255) / 256, 256, 0, stream>>>(W_in, Winb, PROJ_DIM * H_DIM);
  cvt_f32_bf16<<<(H_DIM * D_SSM + 255) / 256, 256, 0, stream>>>(W_out, Woutb, H_DIM * D_SSM);
  rmsnorm_bf16<<<L_SEQ, 256, 0, stream>>>(x, rms_w, hb);
  // in_proj: proj[l,p] = sum_k h[l,k] * W_in[p,k]   (M=2048, N=8512, K=2048)
  gemm_bf16_abt<<<dim3(PROJ_DIM / 64, L_SEQ / 256, 1), 256, 0, stream>>>(
      hb, Winb, proj, L_SEQ, PROJ_DIM, H_DIM, H_DIM, H_DIM, PROJ_DIM, 0, 0, 0, nullptr, 0);
  conv_silu<<<(L_SEQ * CONV_DIM + 255) / 256, 256, 0, stream>>>(proj, cw, cb, xc, xsb, Bmb, Cmb);
  dt_softplus<<<(L_SEQ * NH + 255) / 256, 256, 0, stream>>>(proj, dtb, dts);
  dA_cumsum<<<NC * NH, CS, 0, stream>>>(dts, A_log, dAc);
  // CB[c][l][s] = sum_n Cm[l,n] * Bm[s,n]  (batched over chunks; M=N=256, K=128)
  gemm_bf16_abt<<<dim3(CS / 64, 1, NC), 256, 0, stream>>>(
      Cmb, Bmb, CBuf, CS, CS, NS, NS, NS, CS,
      (long)CS * NS, (long)CS * NS, (long)CS * CS, nullptr, 0);
  ydiag_wmma<<<dim3(NH, NC), 256, 0, stream>>>(CBuf, dAc, dts, xsb, xc, Dv, ybuf);
  states_kernel<<<dim3(NH, NC), 256, 0, stream>>>(xc, dAc, dts, st);
  chunk_scan<<<(NH * PD * NS) / 256, 256, 0, stream>>>(st, dAc, pst);
  yoff_kernel<<<dim3(NH, NC), 256, 0, stream>>>(xc, dAc, pst, ybuf);
  gate_norm<<<L_SEQ, 256, 0, stream>>>(ybuf, proj, nw, ynb);
  // out_proj + residual: out[l,h] = x[l,h] + sum_d y[l,d] * W_out[h,d]  (M=2048, N=2048, K=4096)
  gemm_bf16_abt<<<dim3(H_DIM / 64, L_SEQ / 256, 1), 256, 0, stream>>>(
      ynb, Woutb, out, L_SEQ, H_DIM, D_SSM, D_SSM, D_SSM, H_DIM, 0, 0, 0, x, H_DIM);
}
